// MambaClassifier_33509334844068
// MI455X (gfx1250) — matrix operand order repaired and checked by reading, not compiled, not run on hardware
//
#include <hip/hip_runtime.h>

typedef _Float16 half_t;
typedef __attribute__((ext_vector_type(16))) _Float16 v16h;
typedef __attribute__((ext_vector_type(8)))  _Float16 v8h;
typedef __attribute__((ext_vector_type(8)))  float    v8f;
typedef __attribute__((ext_vector_type(4)))  int      i32x4;

#define D_MODEL 128
#define D_INNER 256
#define D_STATE 16
#define DT_RANK 8
#define NLAYERS 4
#define SEQ_B   64
#define SEQ_L   1024
#define MTOT    (SEQ_B * SEQ_L)
#define DBL_N   48   // DT_RANK + 2*D_STATE = 40, padded to 3 WMMA tiles

// CDNA5 async global->LDS path, guarded so the file always compiles.
#if __has_builtin(__builtin_amdgcn_global_load_async_to_lds_b128) && \
    __has_builtin(__builtin_amdgcn_s_wait_asynccnt)
#define HAVE_ASYNC_LDS 1
typedef __attribute__((address_space(1))) i32x4* gas_v4i;   // global, 16B chunk
typedef __attribute__((address_space(3))) i32x4* las_v4i;   // LDS, 16B chunk
#else
#define HAVE_ASYNC_LDS 0
#endif

// ---------------------------------------------------------------------------
// f16 WMMA GEMM: C[M,N] = A[M,K] * W[N,K]^T  (f32 accumulate)
// Block = 256 threads = 8 waves = 8 consecutive M tiles; block covers NT*16
// columns. The W panel (NT*16 x K) is staged once per block into LDS (padded
// row stride K+8 halves to spread ds_load_b128 across banks); activations
// stream from global with prefetch.
// ---------------------------------------------------------------------------
template<int NT, int K>
__global__ void __launch_bounds__(256) wmma_gemm_kernel(
    const half_t* __restrict__ A, const half_t* __restrict__ Bw,
    float* __restrict__ C, half_t* __restrict__ Ch,
    int M, int N)
{
    constexpr int LDB = K + 8;                 // padded LDS row stride (halves)
    __shared__ half_t sB[NT * 16 * LDB];

    const int tid  = threadIdx.x;
    const int lane = tid & 31;
    const int wave = tid >> 5;
    const int m0   = (blockIdx.x * 8 + wave) * 16;
    const int n0   = blockIdx.y * (NT * 16);

    // ---- cooperative stage of the B panel into LDS (16B chunks) ----
    constexpr int CHUNKS = NT * 16 * (K / 8);  // multiples of 256 for all uses
    for (int idx = tid; idx < CHUNKS; idx += 256) {
        const int row = idx / (K / 8);
        const int c8  = (idx - row * (K / 8)) * 8;
        half_t* src = const_cast<half_t*>(Bw) + (size_t)(n0 + row) * K + c8;
        half_t* dst = &sB[row * LDB + c8];
#if HAVE_ASYNC_LDS
        __builtin_amdgcn_global_load_async_to_lds_b128((gas_v4i)src, (las_v4i)dst, 0, 0);
#else
        *(v8h*)dst = *(const v8h*)src;
#endif
    }
#if HAVE_ASYNC_LDS
    __builtin_amdgcn_s_wait_asynccnt(0);
#endif
    __syncthreads();
    if (m0 >= M) return;                       // wave-uniform (never taken here)

    const int half16 = lane >> 4;              // 0 or 1
    const int l15    = lane & 15;
    const int sel8   = half16 * 8;
    const int sel16  = half16 * 16;
    const half_t* Arow = A + (size_t)(m0 + l15) * K;

    v8f acc[NT];
#pragma unroll
    for (int t = 0; t < NT; ++t)
#pragma unroll
        for (int i = 0; i < 8; ++i) acc[t][i] = 0.0f;

#pragma unroll
    for (int k0 = 0; k0 < K; k0 += 32) {
        __builtin_prefetch(Arow + k0 + 32, 0, 1);      // global_prefetch_b8
        // A fragment: 16x32 f16 per ISA layout
        const v8h lo = *(const v8h*)(Arow + k0 + sel8);
        const v8h hi = *(const v8h*)(Arow + k0 + 16 + sel8);
        v16h a;
#pragma unroll
        for (int j = 0; j < 8; ++j) { a[j] = lo[j]; a[j + 8] = hi[j]; }

        // gather ALL B fragments from LDS first, then issue the WMMAs
        v16h bfrag[NT];
#pragma unroll
        for (int t = 0; t < NT; ++t) {
            const half_t* bp = &sB[(t * 16 + l15) * LDB + k0 + sel8];
            const v8h b0 = *(const v8h*)bp;
            const v8h b1 = *(const v8h*)(bp + 16);
#pragma unroll
            for (int j = 0; j < 8; ++j) { bfrag[t][j] = b0[j]; bfrag[t][j + 8] = b1[j]; }
        }
#pragma unroll
        for (int t = 0; t < NT; ++t)
            acc[t] = __builtin_amdgcn_wmma_f32_16x16x32_f16(
                         false, a, false, bfrag[t], (short)0, acc[t], false, false);
    }

    // C/D layout: lane -> col (l15); VGPR i -> row sel8 + i
#pragma unroll
    for (int t = 0; t < NT; ++t) {
        const int cn = n0 + t * 16 + l15;
#pragma unroll
        for (int i = 0; i < 8; ++i) {
            const int cm = m0 + sel8 + i;
            const size_t idx = (size_t)cm * N + cn;
            C[idx] = acc[t][i];
            if (Ch) Ch[idx] = (half_t)acc[t][i];
        }
    }
}

// h[m,c] = x[m] * Wp[c] + bp[c]   (C=1 input projection), plus f16 shadow
__global__ void embed_kernel(const float* __restrict__ x, const float* __restrict__ Wp,
                             const float* __restrict__ bp, float* __restrict__ h,
                             half_t* __restrict__ h16)
{
    const int m = blockIdx.x;
    const int c = threadIdx.x;   // 128
    const float v = x[m] * Wp[c] + bp[c];
    h[(size_t)m * D_MODEL + c]   = v;
    h16[(size_t)m * D_MODEL + c] = (half_t)v;
}

__global__ void f32_to_f16_kernel(const float* __restrict__ s, half_t* __restrict__ d, int n)
{
    const int i = blockIdx.x * blockDim.x + threadIdx.x;
    if (i < n) d[i] = (half_t)s[i];
}

// Stage Wx (40 x 256) into f16 padded to (48 x 256) with zero rows 40..47
__global__ void pad_wx_kernel(const float* __restrict__ Wx, half_t* __restrict__ d)
{
    const int i = blockIdx.x * blockDim.x + threadIdx.x;
    if (i >= DBL_N * D_INNER) return;
    const int row = i / D_INNER, k = i - row * D_INNER;
    d[i] = (half_t)((row < (DT_RANK + 2 * D_STATE)) ? Wx[row * D_INNER + k] : 0.0f);
}

// Depthwise causal conv (taps 4) over the u-half of xz, + bias + SiLU -> f16
__global__ void __launch_bounds__(256) conv_silu_kernel(
    const float* __restrict__ xz, const float* __restrict__ Wconv,
    const float* __restrict__ bconv, half_t* __restrict__ u16)
{
    const int m = blockIdx.x;       // 0..MTOT-1
    const int d = threadIdx.x;      // 0..255
    const int l = m & (SEQ_L - 1);
    const float w0 = Wconv[d * 4 + 0], w1 = Wconv[d * 4 + 1];
    const float w2 = Wconv[d * 4 + 2], w3 = Wconv[d * 4 + 3];
    const float* base = xz + (size_t)m * (2 * D_INNER) + d;
    float acc = bconv[d] + w3 * base[0];
    if (l >= 1) acc += w2 * base[-1 * (2 * D_INNER)];
    if (l >= 2) acc += w1 * base[-2 * (2 * D_INNER)];
    if (l >= 3) acc += w0 * base[-3 * (2 * D_INNER)];
    const float s = acc / (1.0f + __expf(-acc));   // SiLU
    u16[(size_t)m * D_INNER + d] = (half_t)s;
}

// Selective scan: block b, lane d; 16-state recurrence in registers.
// Fuses delta = softplus(dt@Wdt + bdt), +u*Dp skip, and SiLU(z) gating.
// dt/B/C (40 floats/step) staged in LDS 8 steps at a time.
__global__ void __launch_bounds__(256) scan_kernel(
    const float* __restrict__ xz, const half_t* __restrict__ u16,
    const float* __restrict__ dbl,
    const float* __restrict__ Wdt, const float* __restrict__ bdt,
    const float* __restrict__ A_log, const float* __restrict__ Dp,
    half_t* __restrict__ y16)
{
    const int b = blockIdx.x;
    const int d = threadIdx.x;
    __shared__ float s_blk[8 * DBL_N];   // 8 steps x 48

    float wdt[DT_RANK];
#pragma unroll
    for (int j = 0; j < DT_RANK; ++j) wdt[j] = Wdt[d * DT_RANK + j];
    const float bdtd = bdt[d];
    float Aa[D_STATE];
#pragma unroll
    for (int n = 0; n < D_STATE; ++n) Aa[n] = -__expf(A_log[d * D_STATE + n]);
    const float Dpd = Dp[d];
    float hs[D_STATE];
#pragma unroll
    for (int n = 0; n < D_STATE; ++n) hs[n] = 0.0f;

    for (int lc = 0; lc < SEQ_L; lc += 8) {
        const size_t dbase = ((size_t)b * SEQ_L + lc) * DBL_N;
        for (int idx = d; idx < 8 * DBL_N; idx += 256) s_blk[idx] = dbl[dbase + idx];
        __syncthreads();
#pragma unroll 1
        for (int s = 0; s < 8; ++s) {
            const float* row = s_blk + s * DBL_N;
            float dtv = bdtd;
#pragma unroll
            for (int j = 0; j < DT_RANK; ++j) dtv += row[j] * wdt[j];
            const float delta = (dtv > 20.0f) ? dtv : __logf(1.0f + __expf(dtv));
            const size_t mi = (size_t)b * SEQ_L + lc + s;
            const float ut = (float)u16[mi * D_INNER + d];
            const float du = delta * ut;
            float accv = 0.0f;
#pragma unroll
            for (int n = 0; n < D_STATE; ++n) {
                hs[n] = __expf(delta * Aa[n]) * hs[n] + du * row[DT_RANK + n];
                accv += hs[n] * row[DT_RANK + D_STATE + n];
            }
            const float z = xz[mi * (2 * D_INNER) + D_INNER + d];
            const float g = z / (1.0f + __expf(-z));
            y16[mi * D_INNER + d] = (half_t)((accv + ut * Dpd) * g);
        }
        __syncthreads();
    }
}

// m[b,c] = mean over L of h[b,l,c]
__global__ void mean_kernel(const float* __restrict__ h, float* __restrict__ m)
{
    const int b = blockIdx.x;
    const int c = threadIdx.x;   // 128
    float s = 0.0f;
    for (int l = 0; l < SEQ_L; ++l)
        s += h[((size_t)b * SEQ_L + l) * D_MODEL + c];
    m[b * D_MODEL + c] = s * (1.0f / (float)SEQ_L);
}

// LayerNorm over 128 + classifier head -> out[b, 10]
__global__ void head_kernel(const float* __restrict__ m, const float* __restrict__ g,
                            const float* __restrict__ bln, const float* __restrict__ Wc,
                            const float* __restrict__ bc, float* __restrict__ out)
{
    __shared__ float sbuf[D_MODEL];
    __shared__ float smn[D_MODEL];
    const int b = blockIdx.x;
    const int c = threadIdx.x;   // 128
    const float v = m[b * D_MODEL + c];
    sbuf[c] = v; __syncthreads();
    for (int s = 64; s > 0; s >>= 1) { if (c < s) sbuf[c] += sbuf[c + s]; __syncthreads(); }
    const float mu = sbuf[0] * (1.0f / D_MODEL);
    __syncthreads();
    const float dv = v - mu;
    sbuf[c] = dv * dv; __syncthreads();
    for (int s = 64; s > 0; s >>= 1) { if (c < s) sbuf[c] += sbuf[c + s]; __syncthreads(); }
    const float var = sbuf[0] * (1.0f / D_MODEL);
    const float mn = dv * rsqrtf(var + 1e-5f) * g[c] + bln[c];
    smn[c] = mn; __syncthreads();
    if (c < 10) {
        float a = bc[c];
        for (int k = 0; k < D_MODEL; ++k) a += smn[k] * Wc[c * D_MODEL + k];
        out[b * 10 + c] = a;
    }
}

extern "C" void kernel_launch(void* const* d_in, const int* in_sizes, int n_in,
                              void* d_out, int out_size, void* d_ws, size_t ws_size,
                              hipStream_t stream)
{
    (void)in_sizes; (void)n_in; (void)out_size; (void)ws_size;
    const float* x     = (const float*)d_in[0];
    const float* Wp    = (const float*)d_in[1];
    const float* bp    = (const float*)d_in[2];
    const float* Win   = (const float*)d_in[3];
    const float* Wconv = (const float*)d_in[4];
    const float* bconv = (const float*)d_in[5];
    const float* Wx    = (const float*)d_in[6];
    const float* Wdt   = (const float*)d_in[7];
    const float* bdt   = (const float*)d_in[8];
    const float* A_log = (const float*)d_in[9];
    const float* Dp    = (const float*)d_in[10];
    const float* Wo    = (const float*)d_in[11];
    const float* g_ln  = (const float*)d_in[12];
    const float* b_ln  = (const float*)d_in[13];
    const float* Wc    = (const float*)d_in[14];
    const float* bc    = (const float*)d_in[15];
    float* out = (float*)d_out;

    char* ws = (char*)d_ws;
    size_t off = 0;
    auto alloc = [&](size_t bytes) -> void* {
        void* p = ws + off; off += (bytes + 255) & ~(size_t)255; return p;
    };
    float*  h    = (float*) alloc((size_t)MTOT * D_MODEL * 4);
    half_t* h16  = (half_t*)alloc((size_t)MTOT * D_MODEL * 2);
    float*  xz   = (float*) alloc((size_t)MTOT * 2 * D_INNER * 4);
    half_t* u16  = (half_t*)alloc((size_t)MTOT * D_INNER * 2);
    float*  dbl  = (float*) alloc((size_t)MTOT * DBL_N * 4);
    half_t* y16  = (half_t*)alloc((size_t)MTOT * D_INNER * 2);
    half_t* wstg = (half_t*)alloc((size_t)(2 * D_INNER) * D_MODEL * 2); // max weight
    float*  mbuf = (float*) alloc((size_t)SEQ_B * D_MODEL * 4);

    embed_kernel<<<MTOT, D_MODEL, 0, stream>>>(x, Wp, bp, h, h16);

    for (int i = 0; i < NLAYERS; ++i) {
        // (a) xz = h @ Win^T   (M x 512, K = 128)
        f32_to_f16_kernel<<<(2 * D_INNER * D_MODEL + 255) / 256, 256, 0, stream>>>(
            Win + (size_t)i * 2 * D_INNER * D_MODEL, wstg, 2 * D_INNER * D_MODEL);
        wmma_gemm_kernel<4, 128><<<dim3(MTOT / 128, (2 * D_INNER) / 64), 256, 0, stream>>>(
            h16, wstg, xz, (half_t*)nullptr, MTOT, 2 * D_INNER);

        // (b) depthwise causal conv + SiLU -> u16
        conv_silu_kernel<<<MTOT, D_INNER, 0, stream>>>(
            xz, Wconv + (size_t)i * D_INNER * 4, bconv + i * D_INNER, u16);

        // (c) dbl = u @ Wx^T  (N padded 40 -> 48, K = 256)
        pad_wx_kernel<<<(DBL_N * D_INNER + 255) / 256, 256, 0, stream>>>(
            Wx + (size_t)i * (DT_RANK + 2 * D_STATE) * D_INNER, wstg);
        wmma_gemm_kernel<3, 256><<<dim3(MTOT / 128, 1), 256, 0, stream>>>(
            u16, wstg, dbl, (half_t*)nullptr, MTOT, DBL_N);

        // (d) fused selective scan + softplus(delta) + D-skip + SiLU(z) gate
        scan_kernel<<<SEQ_B, D_INNER, 0, stream>>>(
            xz, u16, dbl,
            Wdt + (size_t)i * D_INNER * DT_RANK, bdt + i * D_INNER,
            A_log + (size_t)i * D_INNER * D_STATE, Dp + i * D_INNER, y16);

        // (e) h = y @ Wo^T  (M x 128, K = 256), write f32 + f16 shadow
        f32_to_f16_kernel<<<(D_MODEL * D_INNER + 255) / 256, 256, 0, stream>>>(
            Wo + (size_t)i * D_MODEL * D_INNER, wstg, D_MODEL * D_INNER);
        wmma_gemm_kernel<4, 256><<<dim3(MTOT / 128, D_MODEL / 64), 256, 0, stream>>>(
            y16, wstg, h, h16, MTOT, D_MODEL);
    }

    mean_kernel<<<SEQ_B, D_MODEL, 0, stream>>>(h, mbuf);
    head_kernel<<<SEQ_B, D_MODEL, 0, stream>>>(mbuf, g_ln, b_ln, Wc, bc, out);
}
